// DoomLiquidNet_61400852463785
// MI455X (gfx1250) — matrix operand order repaired and checked by reading, not compiled, not run on hardware
//
#include <hip/hip_runtime.h>

// ---------------- feature detection: async global->LDS (gfx1250 ASYNCcnt path) ----------------
#if defined(__has_builtin)
# if __has_builtin(__builtin_amdgcn_global_load_async_to_lds_b128) && \
     __has_builtin(__builtin_amdgcn_s_wait_asynccnt)
#  define HAVE_ASYNC_LDS 1
# endif
#endif
#ifndef HAVE_ASYNC_LDS
# define HAVE_ASYNC_LDS 0
#endif

// ---------------- bf16 helpers ----------------
typedef __attribute__((ext_vector_type(16))) __bf16 bf16x16;
typedef __attribute__((ext_vector_type(8)))  float  f32x8;

__device__ __forceinline__ unsigned short f32_bf16_bits(float f) {
    union { float f; unsigned u; } v; v.f = f;
    unsigned r = v.u + 0x7FFFu + ((v.u >> 16) & 1u);   // round-to-nearest-even
    return (unsigned short)(r >> 16);
}
__device__ __forceinline__ __bf16 bits_bf16(unsigned short s) {
    return __builtin_bit_cast(__bf16, s);
}
__device__ __forceinline__ __bf16 f32_bf16(float f) { return bits_bf16(f32_bf16_bits(f)); }

__device__ __forceinline__ f32x8 wmma_bf16(bf16x16 a, bf16x16 b, f32x8 c) {
    // D(f32 16x16) = A(16x32 bf16) * B(32x16 bf16) + C ; emits v_wmma_f32_16x16x32_bf16
    return __builtin_amdgcn_wmma_f32_16x16x32_bf16(false, a, false, b, (short)0, c, false, false);
}

// A-fragment element->K mapping (16-bit A 16x32, ISA 7.12.2): lanes0-15 hold K{0..7,16..23}, hi half +8
__device__ __forceinline__ int a_k_of(int e, int hi) {
    int p = e >> 1;
    return (p < 4 ? 2 * p : 16 + 2 * (p - 4)) + (hi ? 8 : 0) + (e & 1);
}
// B-fragment element->K mapping (16-bit B 32x16): same as A: lanes0-15 hold K{0..7,16..23}, hi half +8
__device__ __forceinline__ int b_k_of(int e, int hi) {
    return (e < 8 ? e : 16 + (e - 8)) + (hi ? 8 : 0);
}

#if HAVE_ASYNC_LDS
// exact builtin parameter types: int __attribute__((vector_size(16))) * in AS1 / AS3
typedef int v4i_gcc __attribute__((vector_size(16)));
typedef __attribute__((address_space(1))) v4i_gcc v4i_global;
typedef __attribute__((address_space(3))) v4i_gcc v4i_lds;

__device__ __forceinline__ void async_copy_b128(const unsigned short* g, unsigned short* l) {
    __builtin_amdgcn_global_load_async_to_lds_b128(
        (v4i_global*)(void*)const_cast<unsigned short*>(g),
        (v4i_lds*)(void*)l, 0, 0);
}
#endif

// ---------------- f32 -> bf16 convert ----------------
__global__ void cvt_f32_bf16(const float* __restrict__ in, unsigned short* __restrict__ out, int n) {
    int i = blockIdx.x * blockDim.x + threadIdx.x;
    if (i < n) out[i] = f32_bf16_bits(in[i]);
}

// ---------------- zero-init f32 buffer ----------------
__global__ void zero_f32(float* __restrict__ p, int n) {
    int i = blockIdx.x * blockDim.x + threadIdx.x;
    if (i < n) p[i] = 0.f;
}

// ---------------- conv1: 3->32, k4 s2 p1, 64x64 -> 32x32, implicit GEMM ----------------
__global__ __launch_bounds__(128) void conv1_wmma(
    const float* __restrict__ x,            // (2048,3,64,64) f32
    const unsigned short* __restrict__ w,   // bf16 (32, 48)  k = c*16+r*4+s
    const float* __restrict__ bias,         // (32,)
    unsigned short* __restrict__ act1)      // bf16 (2048,32,32,32)
{
    const int lane = threadIdx.x & 31;
    const int wave = threadIdx.x >> 5;       // 0..3
    const int img  = blockIdx.x >> 4;        // 2048 images
    const int tile = blockIdx.x & 15;        // 16 tiles x 64 px
    const int m_base = tile * 64 + wave * 16;
    const int hi = lane >> 4;
    const int mr = lane & 15;
    const int op = m_base + mr;
    const int oy = op >> 5, ox = op & 31;
    const float* xin = x + (size_t)img * 3 * 64 * 64;

    bf16x16 a0, a1;
#pragma unroll
    for (int e = 0; e < 16; ++e) {
        int kk = a_k_of(e, hi);
        float f0 = 0.f, f1 = 0.f;
        {   int k = kk;                       // k-tile 0: k in [0,32)
            int c = k >> 4, rr = (k >> 2) & 3, ss = k & 3;
            int iy = 2 * oy - 1 + rr, ix = 2 * ox - 1 + ss;
            if (iy >= 0 && iy < 64 && ix >= 0 && ix < 64)
                f0 = xin[(c * 64 + iy) * 64 + ix];
        }
        {   int k = 32 + kk;                  // k-tile 1: valid for k<48
            if (k < 48) {
                int c = k >> 4, rr = (k >> 2) & 3, ss = k & 3;
                int iy = 2 * oy - 1 + rr, ix = 2 * ox - 1 + ss;
                if (iy >= 0 && iy < 64 && ix >= 0 && ix < 64)
                    f1 = xin[(c * 64 + iy) * 64 + ix];
            }
        }
        a0[e] = f32_bf16(f0);
        a1[e] = f32_bf16(f1);
    }

    for (int n0 = 0; n0 < 32; n0 += 16) {
        const int n = n0 + (lane & 15);
        bf16x16 b0, b1;
#pragma unroll
        for (int e = 0; e < 16; ++e) {
            int k = b_k_of(e, hi);
            b0[e] = bits_bf16(w[n * 48 + k]);
            int k1 = 32 + k;
            b1[e] = (k1 < 48) ? bits_bf16(w[n * 48 + k1]) : bits_bf16((unsigned short)0);
        }
        f32x8 c = {};
        c = wmma_bf16(a0, b0, c);
        c = wmma_bf16(a1, b1, c);
        const float bv = bias[n];
#pragma unroll
        for (int v = 0; v < 8; ++v) {
            int m  = m_base + v + hi * 8;     // output pixel index 0..1023
            float r = c[v] + bv;
            r = r > 0.f ? r : 0.f;
            act1[(size_t)img * 32768 + (size_t)n * 1024 + m] = f32_bf16_bits(r);
        }
    }
}

// ---------------- conv2: 32->64, k4 s2 p1, 32x32 -> 16x16, implicit GEMM ----------------
__global__ __launch_bounds__(128) void conv2_wmma(
    const unsigned short* __restrict__ act1, // bf16 (2048,32,32,32)
    const unsigned short* __restrict__ w,    // bf16 (64, 512) k = c*16+r*4+s
    const float* __restrict__ bias,          // (64,)
    unsigned short* __restrict__ feats)      // bf16 (2048, 16384), idx oc*256+oy*16+ox
{
    const int lane = threadIdx.x & 31;
    const int wave = threadIdx.x >> 5;
    const int img  = blockIdx.x >> 2;        // 2048 images, 4 tiles x 64 px
    const int tile = blockIdx.x & 3;
    const int m_base = tile * 64 + wave * 16;
    const int hi = lane >> 4;
    const int mr = lane & 15;
    const int op = m_base + mr;              // 0..255
    const int oy = op >> 4, ox = op & 15;
    const unsigned short* ain = act1 + (size_t)img * 32768;

    f32x8 acc[4] = {};
    for (int k0 = 0; k0 < 512; k0 += 32) {
        bf16x16 a;
#pragma unroll
        for (int e = 0; e < 16; ++e) {
            int k = k0 + a_k_of(e, hi);
            int c = k >> 4, rr = (k >> 2) & 3, ss = k & 3;
            int iy = 2 * oy - 1 + rr, ix = 2 * ox - 1 + ss;
            unsigned short bits = 0;
            if (iy >= 0 && iy < 32 && ix >= 0 && ix < 32)
                bits = ain[(c * 32 + iy) * 32 + ix];
            a[e] = bits_bf16(bits);
        }
#pragma unroll
        for (int nt = 0; nt < 4; ++nt) {
            const int n = nt * 16 + (lane & 15);
            bf16x16 b;
#pragma unroll
            for (int e = 0; e < 16; ++e) {
                int k = k0 + b_k_of(e, hi);
                b[e] = bits_bf16(w[n * 512 + k]);
            }
            acc[nt] = wmma_bf16(a, b, acc[nt]);
        }
    }
#pragma unroll
    for (int nt = 0; nt < 4; ++nt) {
        const int n = nt * 16 + (lane & 15);
        const float bv = bias[n];
#pragma unroll
        for (int v = 0; v < 8; ++v) {
            int m = m_base + v + hi * 8;     // = oy*16+ox
            float r = acc[nt][v] + bv;
            r = r > 0.f ? r : 0.f;
            feats[(size_t)img * 16384 + (size_t)n * 256 + m] = f32_bf16_bits(r);
        }
    }
}

// ---------------- feats GEMM: xa += feats(2048x16384) @ bb_w[0:16384,0:128] ----------------
// Grid: 32 M-blocks x 4 K-slices. WG = 8 waves (4M x 2N), tile 64(M) x 128(N).
// Double-buffered LDS; async global->LDS copies (3 x b128 per thread per chunk) when available.
// Partial results reduced into xa with global_atomic_add_f32 (xa zeroed beforehand).
__global__ __launch_bounds__(256) void gemm_feats(
    const unsigned short* __restrict__ feats, // bf16 (2048,16384)
    const unsigned short* __restrict__ bbw,   // bf16 (16448,128) row-major
    float* __restrict__ xa)                   // f32 (2048,128), pre-zeroed
{
    __shared__ unsigned short sA[2][64 * 32];   // [buf][m][k]  4KB each
    __shared__ unsigned short sB[2][32 * 128];  // [buf][k][n]  8KB each
    const int tid  = threadIdx.x;
    const int lane = tid & 31;
    const int wave = tid >> 5;
    const int wm = wave >> 1;                 // 0..3 : M sub-tile
    const int wn = wave & 1;                  // 0..1 : N half (64 cols)
    const int mblk   = blockIdx.x >> 2;       // 0..31
    const int kslice = blockIdx.x & 3;        // 0..3
    const int m0 = mblk * 64;
    const int kbase = kslice * 4096;          // 128 chunks of 32
    const int hi = lane >> 4;
    const int mr = lane & 15;
    const int col = lane & 15;

    // per-thread staging coordinates (16B = 8 bf16 per load)
    const int am = tid >> 2;                  // 0..63 : A row
    const int ak = (tid & 3) * 8;             // A k-offset
    const int bk = tid >> 4;                  // 0..15 : B k (first half)
    const int bn = (tid & 15) * 8;            // B n-offset

    f32x8 acc[4] = {};

#if HAVE_ASYNC_LDS
#define STAGE_ONE(G, L) async_copy_b128((G), (L))
#else
#define STAGE_ONE(G, L) (*(uint4*)(L) = *(const uint4*)(G))
#endif

#define STAGE_CHUNK(K0, BUF)                                                          \
    do {                                                                              \
        const unsigned short* gA = feats + (size_t)(m0 + am) * 16384 + (K0) + ak;     \
        const unsigned short* gB0 = bbw + (size_t)((K0) + bk) * 128 + bn;             \
        const unsigned short* gB1 = bbw + (size_t)((K0) + 16 + bk) * 128 + bn;        \
        STAGE_ONE(gA,  &sA[BUF][am * 32 + ak]);                                       \
        STAGE_ONE(gB0, &sB[BUF][bk * 128 + bn]);                                      \
        STAGE_ONE(gB1, &sB[BUF][(16 + bk) * 128 + bn]);                               \
    } while (0)

    STAGE_CHUNK(kbase, 0);

    for (int it = 0; it < 128; ++it) {
        const int buf = it & 1;
        if (it + 1 < 128) {
            STAGE_CHUNK(kbase + (it + 1) * 32, buf ^ 1);
#if HAVE_ASYNC_LDS
            __builtin_amdgcn_s_wait_asynccnt(3);   // current chunk landed; next stays in flight
#endif
        } else {
#if HAVE_ASYNC_LDS
            __builtin_amdgcn_s_wait_asynccnt(0);
#endif
        }
        __syncthreads();

        bf16x16 a;
#pragma unroll
        for (int e = 0; e < 16; ++e)
            a[e] = bits_bf16(sA[buf][(wm * 16 + mr) * 32 + a_k_of(e, hi)]);
#pragma unroll
        for (int nt = 0; nt < 4; ++nt) {
            bf16x16 b;
#pragma unroll
            for (int e = 0; e < 16; ++e)
                b[e] = bits_bf16(sB[buf][b_k_of(e, hi) * 128 + wn * 64 + nt * 16 + col]);
            acc[nt] = wmma_bf16(a, b, acc[nt]);
        }
        __syncthreads();
    }
#undef STAGE_ONE
#undef STAGE_CHUNK

#pragma unroll
    for (int nt = 0; nt < 4; ++nt) {
#pragma unroll
        for (int v = 0; v < 8; ++v) {
            int m = m0 + wm * 16 + v + hi * 8;
            int n = wn * 64 + nt * 16 + col;
            atomicAdd(&xa[(size_t)m * 128 + n], acc[nt][v]);
        }
    }
}

// ---------------- recurrent CfC core: one workgroup, 64 sequential steps ----------------
__global__ __launch_bounds__(256) void cfc_rnn(
    const float* __restrict__ xa,     // (32*64, 128) precomputed x-part of bb pre-activation
    const float* __restrict__ hx0,    // (32,64)
    const float* __restrict__ bbw,    // f32 (16448,128): rows 16384..16447 = h-part
    const float* __restrict__ bb_b,
    const float* __restrict__ ff1_w, const float* __restrict__ ff1_b,
    const float* __restrict__ ff2_w, const float* __restrict__ ff2_b,
    const float* __restrict__ ta_w,  const float* __restrict__ ta_b,
    const float* __restrict__ tb_w,  const float* __restrict__ tb_b,
    const float* __restrict__ out_w, const float* __restrict__ out_b,
    float* __restrict__ logits,       // (32,64,8)
    float* __restrict__ hx_out)       // (32,64)
{
    __shared__ float h[32 * 64];
    __shared__ float bb[32 * 128];
    __shared__ float hw[64 * 128];
    const int tid = threadIdx.x;

    for (int i = tid; i < 2048; i += 256) h[i] = hx0[i];
    for (int i = tid; i < 8192; i += 256) hw[i] = bbw[(size_t)16384 * 128 + i];
    __syncthreads();

    for (int t = 0; t < 64; ++t) {
        // bb = lecun_tanh(xa[:,t,:] + h @ hw + bb_b)
        for (int o = tid; o < 4096; o += 256) {
            int b = o >> 7, j = o & 127;
            float s = xa[((size_t)b * 64 + t) * 128 + j] + bb_b[j];
            const float* hr = h + b * 64;
#pragma unroll 8
            for (int i = 0; i < 64; ++i) s += hr[i] * hw[i * 128 + j];
            bb[o] = 1.7159f * tanhf(0.666f * s);
        }
        __syncthreads();
        // h_new = tanh(bb@ff1)*(1-sig) + sig*tanh(bb@ff2), sig = sigmoid(bb@ta + bb@tb)
        for (int o = tid; o < 2048; o += 256) {
            int b = o >> 6, u = o & 63;
            const float* bbr = bb + b * 128;
            float s1 = ff1_b[u], s2 = ff2_b[u], sa = ta_b[u], sb = tb_b[u];
#pragma unroll 4
            for (int i = 0; i < 128; ++i) {
                float xv = bbr[i];
                s1 += xv * ff1_w[i * 64 + u];
                s2 += xv * ff2_w[i * 64 + u];
                sa += xv * ta_w[i * 64 + u];
                sb += xv * tb_w[i * 64 + u];
            }
            float f1 = tanhf(s1), f2 = tanhf(s2);
            float ti = 1.f / (1.f + __expf(-(sa + sb)));
            h[o] = f1 * (1.f - ti) + ti * f2;
        }
        __syncthreads();
        // logits[:, t, :] = h_new @ out_w + out_b
        for (int o = tid; o < 256; o += 256) {
            int b = o >> 3, c = o & 7;
            float s = out_b[c];
            const float* hr = h + b * 64;
#pragma unroll 8
            for (int u = 0; u < 64; ++u) s += hr[u] * out_w[u * 8 + c];
            logits[((size_t)b * 64 + t) * 8 + c] = s;
        }
        __syncthreads();
    }
    for (int i = tid; i < 2048; i += 256) hx_out[i] = h[i];
}

// ---------------- host launch ----------------
extern "C" void kernel_launch(void* const* d_in, const int* in_sizes, int n_in,
                              void* d_out, int out_size, void* d_ws, size_t ws_size,
                              hipStream_t stream) {
    const float* x_vis   = (const float*)d_in[0];
    const float* hx      = (const float*)d_in[1];
    const float* conv1_w = (const float*)d_in[2];
    const float* conv1_b = (const float*)d_in[3];
    const float* conv2_w = (const float*)d_in[4];
    const float* conv2_b = (const float*)d_in[5];
    const float* bb_w    = (const float*)d_in[6];
    const float* bb_b    = (const float*)d_in[7];
    const float* ff1_w   = (const float*)d_in[8];
    const float* ff1_b   = (const float*)d_in[9];
    const float* ff2_w   = (const float*)d_in[10];
    const float* ff2_b   = (const float*)d_in[11];
    const float* ta_w    = (const float*)d_in[12];
    const float* ta_b    = (const float*)d_in[13];
    const float* tb_w    = (const float*)d_in[14];
    const float* tb_b    = (const float*)d_in[15];
    const float* out_w   = (const float*)d_in[16];
    const float* out_b   = (const float*)d_in[17];

    // workspace layout (~207 MB)
    char* ws = (char*)d_ws;
    const size_t off_act1  = 0;                             // bf16 2048*32*32*32
    const size_t off_feats = off_act1  + (size_t)134217728; // bf16 2048*16384
    const size_t off_bbw   = off_feats + (size_t)67108864;  // bf16 16448*128
    const size_t off_c1w   = off_bbw   + (size_t)4210688;   // bf16 1536
    const size_t off_c2w   = off_c1w   + (size_t)4096;      // bf16 32768
    const size_t off_xa    = off_c2w   + (size_t)65536;     // f32 2048*128
    unsigned short* act1  = (unsigned short*)(ws + off_act1);
    unsigned short* feats = (unsigned short*)(ws + off_feats);
    unsigned short* bbw16 = (unsigned short*)(ws + off_bbw);
    unsigned short* c1w16 = (unsigned short*)(ws + off_c1w);
    unsigned short* c2w16 = (unsigned short*)(ws + off_c2w);
    float*          xa    = (float*)(ws + off_xa);

    // 1) weight converts to bf16 + zero the K-split accumulator
    cvt_f32_bf16<<<(16448 * 128 + 255) / 256, 256, 0, stream>>>(bb_w, bbw16, 16448 * 128);
    cvt_f32_bf16<<<(1536 + 255) / 256, 256, 0, stream>>>(conv1_w, c1w16, 1536);
    cvt_f32_bf16<<<(32768 + 255) / 256, 256, 0, stream>>>(conv2_w, c2w16, 32768);
    zero_f32<<<(2048 * 128 + 255) / 256, 256, 0, stream>>>(xa, 2048 * 128);

    // 2) conv1 implicit GEMM (WMMA bf16)
    conv1_wmma<<<2048 * 16, 128, 0, stream>>>(x_vis, c1w16, conv1_b, act1);

    // 3) conv2 implicit GEMM (WMMA bf16)
    conv2_wmma<<<2048 * 4, 128, 0, stream>>>(act1, c2w16, conv2_b, feats);

    // 4) time-independent backbone GEMM (WMMA bf16, double-buffered async LDS, K-split x4)
    gemm_feats<<<32 * 4, 256, 0, stream>>>(feats, bbw16, xa);

    // 5) sequential CfC core + output head (f32, one workgroup)
    cfc_rnn<<<1, 256, 0, stream>>>(xa, hx, bb_w, bb_b,
                                   ff1_w, ff1_b, ff2_w, ff2_b,
                                   ta_w, ta_b, tb_w, tb_b,
                                   out_w, out_b,
                                   (float*)d_out, (float*)d_out + 32 * 64 * 8);
}
